// GraphTransformerArchitecture_4398046511395
// MI455X (gfx1250) — matrix operand order repaired and checked by reading, not compiled, not run on hardware
//
#include <hip/hip_runtime.h>
#include <hip/hip_bf16.h>

#define N_NODES 50000
#define N_EDGES 800000
#define F_IN 32
#define HID 64

typedef __attribute__((ext_vector_type(16))) _Float16 v16h;
typedef __attribute__((ext_vector_type(8)))  _Float16 v8h;
typedef __attribute__((ext_vector_type(8)))  float    v8f;

// ---------------------------------------------------------------- utilities

__global__ void zero_f32(float* p, int n) {
    int i = blockIdx.x * blockDim.x + threadIdx.x;
    if (i < n) p[i] = 0.0f;
}

__global__ void f32_to_f16(const float* in, _Float16* out, int n) {
    int i = blockIdx.x * blockDim.x + threadIdx.x;
    if (i < n) out[i] = (_Float16)in[i];
}

// Pack up to 4 column-blocks of weights [d x colsEach] (row-major) into the
// WMMA B-operand lane layout:
//   out[((kt*nColTiles + nt)*32 + lane)*16 + j] = W[kt*32 + (lane>>4)*16 + j][col]
__global__ void pack_B(const float* W0, const float* W1, const float* W2, const float* W3,
                       int colsEach, int d, int nCols, _Float16* out) {
    int idx = blockIdx.x * blockDim.x + threadIdx.x;
    int kTiles = d / 32, nColTiles = nCols / 16;
    int total = kTiles * nColTiles * 32 * 16;
    if (idx >= total) return;
    int j    = idx & 15;
    int lane = (idx >> 4) & 31;
    int t    = idx >> 9;            // tile index = kt*nColTiles + nt
    int nt   = t % nColTiles;
    int kt   = t / nColTiles;
    int k    = kt * 32 + (lane >> 4) * 16 + j;
    int col  = nt * 16 + (lane & 15);
    int wi   = col / colsEach;
    int lc   = col % colsEach;
    const float* W = (wi == 0) ? W0 : (wi == 1) ? W1 : (wi == 2) ? W2 : W3;
    float v = (W != nullptr) ? W[(size_t)k * colsEach + lc] : 0.0f;
    out[idx] = (_Float16)v;
}

__global__ void pack_bias(const float* b0, const float* b1, const float* b2, const float* b3,
                          int colsEach, int nCols, float* out) {
    int c = blockIdx.x * blockDim.x + threadIdx.x;
    if (c >= nCols) return;
    int wi = c / colsEach, lc = c % colsEach;
    const float* b = (wi == 0) ? b0 : (wi == 1) ? b1 : (wi == 2) ? b2 : b3;
    out[c] = (b != nullptr) ? b[lc] : 0.0f;
}

// ------------------------------------------------------------ WMMA GEMM
// C[N, nCols] = A[N, dA](f16) x Bpacked(dA x nCols)(f16) + bias, optional relu.
// One wave per 16x16 output tile. N must be a multiple of 16 (50000 = 3125*16).
__global__ __launch_bounds__(256)
void wmma_gemm(const _Float16* __restrict__ A, const _Float16* __restrict__ Bp,
               const float* __restrict__ bias, float* __restrict__ C,
               int N, int dA, int nCols, int doRelu) {
    int wavesPerBlock = blockDim.x >> 5;
    int wid  = blockIdx.x * wavesPerBlock + (threadIdx.x >> 5);
    int lane = threadIdx.x & 31;
    int nColTiles = nCols >> 4;
    int totalTiles = (N >> 4) * nColTiles;
    if (wid >= totalTiles) return;            // uniform per wave
    int rt = wid / nColTiles;
    int nt = wid % nColTiles;
    int m  = lane & 15;
    int hi = lane >> 4;
    int kTiles = dA >> 5;

    const _Float16* Arow = A + (size_t)(rt * 16 + m) * dA;
    v8f c = {};
    for (int kt = 0; kt < kTiles; ++kt) {
        const _Float16* ap = Arow + kt * 32;
        // A lane layout: elems 0..15 -> K = hi*16 + j
        v8h a0 = *(const v8h*)(ap + hi * 16);
        v8h a1 = *(const v8h*)(ap + hi * 16 + 8);
        v16h a = __builtin_shufflevector(a0, a1, 0,1,2,3,4,5,6,7,8,9,10,11,12,13,14,15);
        const _Float16* bp = Bp + ((size_t)(kt * nColTiles + nt) * 32 + lane) * 16;
        v8h b0 = *(const v8h*)(bp);
        v8h b1 = *(const v8h*)(bp + 8);
        v16h b = __builtin_shufflevector(b0, b1, 0,1,2,3,4,5,6,7,8,9,10,11,12,13,14,15);
        c = __builtin_amdgcn_wmma_f32_16x16x32_f16(false, a, false, b, (short)0, c,
                                                   false, false);
    }
    int col = nt * 16 + (lane & 15);
    float bb = bias[col];
    #pragma unroll
    for (int r = 0; r < 8; ++r) {             // C/D layout: row = rt*16 + hi*8 + r
        float v = c[r] + bb;
        if (doRelu) v = fmaxf(v, 0.0f);
        C[(size_t)(rt * 16 + hi * 8 + r) * nCols + col] = v;
    }
}

// ------------------------------------------------------------ LayerNorm(x, 32) wave/row
__global__ void layernorm32(const float* __restrict__ x, const float* g, const float* b,
                            float* __restrict__ out, int N) {
    int row  = blockIdx.x * (blockDim.x >> 5) + (threadIdx.x >> 5);
    int lane = threadIdx.x & 31;
    if (row >= N) return;
    float v = x[(size_t)row * 32 + lane];
    float s = v;
    #pragma unroll
    for (int o = 16; o > 0; o >>= 1) s += __shfl_xor(s, o, 32);
    float mu = s * (1.0f / 32.0f);
    float d = v - mu;
    float var = d * d;
    #pragma unroll
    for (int o = 16; o > 0; o >>= 1) var += __shfl_xor(var, o, 32);
    var *= (1.0f / 32.0f);
    float inv = rsqrtf(var + 1e-5f);
    out[(size_t)row * 32 + lane] = d * inv * g[lane] + b[lane];
}

// ------------------------------------------------------------ edge pass 1
// qkvs layout per node: [0..63]=q, [64..127]=k, [128..191]=v, [192..255]=skip
// One wave32 per edge: lanes load 2 q/k channels each (coalesced), per-head
// dots via 16-lane shuffle reductions; p=exp(score/sqrt(C)); denom atomics.
__global__ void edge_pass1(const float* __restrict__ qkvs, const int* __restrict__ src,
                           const int* __restrict__ dst, int E, int H, int Cd,
                           float* __restrict__ denom, float* __restrict__ pbuf) {
    int e    = blockIdx.x * (blockDim.x >> 5) + (threadIdx.x >> 5);
    int lane = threadIdx.x & 31;
    if (e >= E) return;
    int s = src[e], d = dst[e];
    const float* q = qkvs + (size_t)d * 256;
    const float* k = qkvs + (size_t)s * 256 + 64;
    float p0 = q[lane]      * k[lane];        // channel lane      (heads 0/1 if Cd=16)
    float p1 = q[lane + 32] * k[lane + 32];   // channel lane+32   (heads 2/3 if Cd=16)
    float rs = rsqrtf((float)Cd);
    if (Cd == 16) {                           // H == 4
        #pragma unroll
        for (int o = 8; o > 0; o >>= 1) {     // reduce within 16-lane halves
            p0 += __shfl_xor(p0, o, 32);
            p1 += __shfl_xor(p1, o, 32);
        }
        if ((lane & 15) == 0) {               // lane 0 -> heads 0,2 ; lane 16 -> heads 1,3
            int h0 = lane >> 4;
            float e0 = expf(p0 * rs), e1 = expf(p1 * rs);
            pbuf[(size_t)e * 4 + h0]     = e0;
            pbuf[(size_t)e * 4 + 2 + h0] = e1;
            atomicAdd(&denom[(size_t)d * 4 + h0],     e0);
            atomicAdd(&denom[(size_t)d * 4 + 2 + h0], e1);
        }
    } else {                                  // H == 1, Cd == 64
        float p = p0 + p1;
        #pragma unroll
        for (int o = 16; o > 0; o >>= 1) p += __shfl_xor(p, o, 32);
        if (lane == 0) {
            float e0 = expf(p * rs);
            pbuf[e] = e0;
            atomicAdd(&denom[d], e0);
        }
    }
}

// ------------------------------------------------------------ edge pass 2
// One thread per (edge, channel): 64 consecutive lanes hit 64 consecutive
// agg addresses -> coalesced v loads and cacheline-coalesced f32 atomics.
__global__ void edge_pass2(const float* __restrict__ qkvs, const int* __restrict__ src,
                           const int* __restrict__ dst, int E, int H, int Cd,
                           const float* __restrict__ denom, const float* __restrict__ pbuf,
                           float* __restrict__ agg) {
    long long t = (long long)blockIdx.x * blockDim.x + threadIdx.x;
    if (t >= (long long)E * 64) return;
    int e = (int)(t >> 6);
    int c = (int)(t & 63);
    int s = src[e], d = dst[e];
    int h = c / Cd;
    float alpha = pbuf[(size_t)e * H + h] / (denom[(size_t)d * H + h] + 1e-16f);
    float v = qkvs[(size_t)s * 256 + 128 + c];
    atomicAdd(&agg[(size_t)d * 64 + c], v * alpha);
}

// ------------------------------------------------------------ layer epilogue
// h_out = maybe_relu( LN(agg + skip) + maybe 0.1*h_in ), wave32 per row (64 feats)
__global__ void finish_layer(const float* __restrict__ agg, const float* __restrict__ qkvs,
                             const float* __restrict__ h_in, const float* g, const float* b,
                             int N, int addRes, int doRelu, float* __restrict__ h_out) {
    int row  = blockIdx.x * (blockDim.x >> 5) + (threadIdx.x >> 5);
    int lane = threadIdx.x & 31;
    if (row >= N) return;
    float v0 = agg[(size_t)row * 64 + lane]      + qkvs[(size_t)row * 256 + 192 + lane];
    float v1 = agg[(size_t)row * 64 + 32 + lane] + qkvs[(size_t)row * 256 + 224 + lane];
    float s = v0 + v1;
    #pragma unroll
    for (int o = 16; o > 0; o >>= 1) s += __shfl_xor(s, o, 32);
    float mu = s * (1.0f / 64.0f);
    float d0 = v0 - mu, d1 = v1 - mu;
    float var = d0 * d0 + d1 * d1;
    #pragma unroll
    for (int o = 16; o > 0; o >>= 1) var += __shfl_xor(var, o, 32);
    var *= (1.0f / 64.0f);
    float inv = rsqrtf(var + 1e-5f);
    float o0 = d0 * inv * g[lane]      + b[lane];
    float o1 = d1 * inv * g[32 + lane] + b[32 + lane];
    if (addRes) {
        o0 += 0.1f * h_in[(size_t)row * 64 + lane];
        o1 += 0.1f * h_in[(size_t)row * 64 + 32 + lane];
    }
    if (doRelu) { o0 = fmaxf(o0, 0.0f); o1 = fmaxf(o1, 0.0f); }
    h_out[(size_t)row * 64 + lane]      = o0;
    h_out[(size_t)row * 64 + 32 + lane] = o1;
}

// ------------------------------------------------------------ MLP heads final
// rt1[n] = [ relu(h@Wr1+br1) | relu(h@Wt1+bt1) ]  (computed by wmma_gemm, relu'd)
__global__ void head_final(const float* __restrict__ rt1,
                           const float* __restrict__ Wr2, const float* __restrict__ br2,
                           const float* __restrict__ Wt2, const float* __restrict__ bt2,
                           float* __restrict__ out, int N) {
    int n = blockIdx.x * blockDim.x + threadIdx.x;
    if (n >= N) return;
    const float* r = rt1 + (size_t)n * 64;
    float a = br2[0], t = bt2[0];
    for (int i = 0; i < 32; ++i) { a += r[i] * Wr2[i]; t += r[32 + i] * Wt2[i]; }
    out[(size_t)n * 2 + 0] = a;
    out[(size_t)n * 2 + 1] = t;
}

// ================================================================ launcher

extern "C" void kernel_launch(void* const* d_in, const int* in_sizes, int n_in,
                              void* d_out, int out_size, void* d_ws, size_t ws_size,
                              hipStream_t stream) {
    const int N = N_NODES, E = N_EDGES;
    const float* x     = (const float*)d_in[0];
    const int*   ei    = (const int*)d_in[1];
    const float* ln0_g = (const float*)d_in[2];
    const float* ln0_b = (const float*)d_in[3];
    const int* src = ei;
    const int* dst = ei + E;

    const float *Wq[3], *bq[3], *Wk[3], *bk[3], *Wv[3], *bv[3], *Ws[3], *g[3], *bln[3];
    for (int l = 0; l < 3; ++l) {
        int base = 4 + 9 * l;
        Wq[l]  = (const float*)d_in[base + 0];  bq[l] = (const float*)d_in[base + 1];
        Wk[l]  = (const float*)d_in[base + 2];  bk[l] = (const float*)d_in[base + 3];
        Wv[l]  = (const float*)d_in[base + 4];  bv[l] = (const float*)d_in[base + 5];
        Ws[l]  = (const float*)d_in[base + 6];
        g[l]   = (const float*)d_in[base + 7];  bln[l] = (const float*)d_in[base + 8];
    }
    const float* Wr1 = (const float*)d_in[31]; const float* br1 = (const float*)d_in[32];
    const float* Wr2 = (const float*)d_in[33]; const float* br2 = (const float*)d_in[34];
    const float* Wt1 = (const float*)d_in[35]; const float* bt1 = (const float*)d_in[36];
    const float* Wt2 = (const float*)d_in[37]; const float* bt2 = (const float*)d_in[38];
    float* out = (float*)d_out;

    // ---- workspace carve-out
    size_t off = 0;
    auto carve = [&](size_t bytes) -> void* {
        void* p = (char*)d_ws + off;
        off += (bytes + 255) & ~(size_t)255;
        return p;
    };
    float*    hA    = (float*)   carve((size_t)N * 64 * 4);
    float*    hB    = (float*)   carve((size_t)N * 64 * 4);
    _Float16* hh    = (_Float16*)carve((size_t)N * 64 * 2);
    float*    qkvs  = (float*)   carve((size_t)N * 256 * 4);
    float*    pbuf  = (float*)   carve((size_t)E * 4 * 4);
    float*    denom = (float*)   carve((size_t)N * 4 * 4);
    float*    agg   = (float*)   carve((size_t)N * 64 * 4);
    _Float16* wpack = (_Float16*)carve((size_t)64 * 256 * 2);
    float*    bcat  = (float*)   carve((size_t)256 * 4);

    const int TPB = 256;
    auto blocksFor = [](long long n, int t) { return (int)((n + t - 1) / t); };

    // 1) input LayerNorm: x[N,32] -> hA (stride 32)
    layernorm32<<<blocksFor((long long)N * 32, TPB), TPB, 0, stream>>>(x, ln0_g, ln0_b, hA, N);

    float* hcur = hA;
    float* hnxt = hB;
    const int Hs[3] = {4, 4, 1};
    const int Cs[3] = {16, 16, 64};

    for (int l = 0; l < 3; ++l) {
        int dA = (l == 0) ? F_IN : HID;
        int H = Hs[l], Cd = Cs[l];

        // convert activations to f16
        f32_to_f16<<<blocksFor((long long)N * dA, TPB), TPB, 0, stream>>>(hcur, hh, N * dA);

        // pack Wq|Wk|Wv|Ws -> wpack (dA x 256), bias q|k|v|0
        int packTotal = (dA / 32) * (256 / 16) * 32 * 16;
        pack_B<<<blocksFor(packTotal, TPB), TPB, 0, stream>>>(
            Wq[l], Wk[l], Wv[l], Ws[l], 64, dA, 256, wpack);
        pack_bias<<<1, 256, 0, stream>>>(bq[l], bk[l], bv[l], nullptr, 64, 256, bcat);

        // zero denom / agg
        zero_f32<<<blocksFor((long long)N * 4, TPB), TPB, 0, stream>>>(denom, N * 4);
        zero_f32<<<blocksFor((long long)N * 64, TPB), TPB, 0, stream>>>(agg, N * 64);

        // fused QKVS projection: [N,dA] x [dA,256]
        {
            int tiles = (N / 16) * (256 / 16);
            wmma_gemm<<<blocksFor(tiles, 8), 256, 0, stream>>>(hh, wpack, bcat, qkvs,
                                                               N, dA, 256, 0);
        }

        // attention: wave-per-edge scores, (edge,channel)-parallel aggregation
        edge_pass1<<<blocksFor((long long)E * 32, TPB), TPB, 0, stream>>>(
            qkvs, src, dst, E, H, Cd, denom, pbuf);
        edge_pass2<<<blocksFor((long long)E * 64, TPB), TPB, 0, stream>>>(
            qkvs, src, dst, E, H, Cd, denom, pbuf, agg);

        // epilogue: +skip, LayerNorm, +0.1*res (l>0), relu (l<2)
        finish_layer<<<blocksFor((long long)N * 32, TPB), TPB, 0, stream>>>(
            agg, qkvs, hcur, g[l], bln[l], N, (l > 0) ? 1 : 0, (l < 2) ? 1 : 0, hnxt);

        float* t = hcur; hcur = hnxt; hnxt = t;
    }

    // 2) heads: rt1 = relu( h @ [Wr1|Wt1] + [br1|bt1] )  via WMMA, then 32-dots
    f32_to_f16<<<blocksFor((long long)N * 64, TPB), TPB, 0, stream>>>(hcur, hh, N * 64);
    {
        int packTotal = (64 / 32) * (64 / 16) * 32 * 16;
        pack_B<<<blocksFor(packTotal, TPB), TPB, 0, stream>>>(
            Wr1, Wt1, nullptr, nullptr, 32, 64, 64, wpack);
        pack_bias<<<1, 64, 0, stream>>>(br1, bt1, nullptr, nullptr, 32, 64, bcat);
        int tiles = (N / 16) * (64 / 16);
        float* rt1 = agg;   // reuse
        wmma_gemm<<<blocksFor(tiles, 8), 256, 0, stream>>>(hh, wpack, bcat, rt1, N, 64, 64, 1);
        head_final<<<blocksFor(N, TPB), TPB, 0, stream>>>(rt1, Wr2, br2, Wt2, bt2, out, N);
    }
}
